// SoftDecisionTree_58325655879930
// MI455X (gfx1250) — matrix operand order repaired and checked by reading, not compiled, not run on hardware
//
#include <hip/hip_runtime.h>

// ---------- types ----------
typedef __attribute__((ext_vector_type(16))) __bf16 v16bf;
typedef __attribute__((ext_vector_type(8)))  float  v8f;
typedef __attribute__((ext_vector_type(4)))  float  f32x4;
typedef __attribute__((ext_vector_type(8)))  __bf16 bf16x8;
typedef __attribute__((ext_vector_type(4)))  unsigned int u32x4;
typedef __attribute__((ext_vector_type(8)))  int i32x8;
typedef __attribute__((ext_vector_type(4)))  int i32x4;

union BF16x16 { v16bf v; bf16x8 h[2]; __bf16 e[16]; };

#if __has_builtin(__builtin_amdgcn_tensor_load_to_lds) && __has_builtin(__builtin_amdgcn_s_wait_tensorcnt)
#define HAVE_TDM 1
#else
#define HAVE_TDM 0
#endif

// ---------- bf16 helpers (round-to-nearest-even split) ----------
__device__ __forceinline__ __bf16 f2bf(float x) {
    unsigned u = __builtin_bit_cast(unsigned, x);
    unsigned r = u + 0x7FFFu + ((u >> 16) & 1u);
    unsigned short h = (unsigned short)(r >> 16);
    return __builtin_bit_cast(__bf16, h);
}
__device__ __forceinline__ float bf2f(__bf16 b) {
    unsigned short h = __builtin_bit_cast(unsigned short, b);
    unsigned u = ((unsigned)h) << 16;
    return __builtin_bit_cast(float, u);
}
__device__ __forceinline__ v8f zero8() {
    v8f z;
#pragma unroll
    for (int i = 0; i < 8; ++i) z[i] = 0.0f;
    return z;
}
__device__ __forceinline__ v8f wmma_bf16(v16bf a, v16bf b, v8f c) {
    return __builtin_amdgcn_wmma_f32_16x16x32_bf16(false, a, false, b,
                                                   (short)0, c, false, false);
}

#if HAVE_TDM
// TDM 2D tile load: tile_d0 x tile_d1 elements (2-byte), row stride stride0
// elements, from gbase into LDS offset lds_off. D# per CDNA5 ISA 08 section 8.
// This toolchain's builtin takes 6 args:
//   (u32x4 group0, i32x8 group1, i32x4 group2, i32x4 group3, i32x8 extra, i32 cpol)
__device__ __forceinline__ void tdm_load_2d_b16(const void* gbase, unsigned lds_off,
                                                unsigned tile_d0, unsigned tile_d1,
                                                unsigned tensor_d0, unsigned tensor_d1,
                                                unsigned stride0) {
    unsigned long long ga = (unsigned long long)(uintptr_t)gbase;
    u32x4 g0;
    g0[0] = 1u;                                            // count=1, user mode
    g0[1] = lds_off;                                       // lds_addr[31:0]
    g0[2] = (unsigned)(ga & 0xFFFFFFFFu);                  // global_addr[31:0]
    g0[3] = (unsigned)((ga >> 32) & 0x01FFFFFFu) | (2u << 30); // ga[56:32] | type=2
    i32x8 g1;
    g1[0] = (int)(1u << 16);                               // data_size=1 (2 bytes)
    g1[1] = (int)((tensor_d0 & 0xFFFFu) << 16);            // tensor_dim0[15:0]
    g1[2] = (int)(((tensor_d0 >> 16) & 0xFFFFu) | ((tensor_d1 & 0xFFFFu) << 16));
    g1[3] = (int)(((tensor_d1 >> 16) & 0xFFFFu) | ((tile_d0 & 0xFFFFu) << 16));
    g1[4] = (int)(tile_d1 & 0xFFFFu);                      // tile_dim1, tile_dim2=0
    g1[5] = (int)stride0;                                  // tensor_dim0_stride[31:0]
    g1[6] = 0;
    g1[7] = 0;
    i32x4 gz;
    gz[0] = 0; gz[1] = 0; gz[2] = 0; gz[3] = 0;
    i32x8 gz8;
#pragma unroll
    for (int i = 0; i < 8; ++i) gz8[i] = 0;
    __builtin_amdgcn_tensor_load_to_lds(g0, g1, gz, gz, gz8, 0);
}
__device__ __forceinline__ unsigned lds_off_u32(const void* p) {
    return (unsigned)(uintptr_t)p;   // generic shared ptr: low 32 bits = LDS offset
}
#endif

// ============================================================
// Kernel A: W (127x2048 f32) -> Whi/Wlo bf16 (128x2048, row 127 = 0)
// ============================================================
__global__ __launch_bounds__(256) void sdt_wconv(const float* __restrict__ W,
                                                 __bf16* __restrict__ Whi,
                                                 __bf16* __restrict__ Wlo) {
    int idx = blockIdx.x * 256 + threadIdx.x;   // 0 .. 128*2048-1
    int r = idx >> 11;
    float v = (r < 127) ? W[idx] : 0.0f;        // flat index valid iff r<127
    __bf16 h = f2bf(v);
    Whi[idx] = h;
    Wlo[idx] = f2bf(v - bf2f(h));
}

// ============================================================
// Kernel B: distsT = softmax(leaf_params, axis=-1) transposed, split hi/lo
// leaf_params: 128 x 1000 f32 ; dT_*: 1024 x 128 bf16 (rows >=1000 zero)
// ============================================================
__global__ __launch_bounds__(256) void sdt_softmax_t(const float* __restrict__ lp,
                                                     __bf16* __restrict__ dT_hi,
                                                     __bf16* __restrict__ dT_lo) {
    __shared__ float red[256];
    const int leaf = blockIdx.x;
    const int tid  = threadIdx.x;
    const float* row = lp + (size_t)leaf * 1000;

    float lmax = -3.4e38f;
    for (int c = tid; c < 1000; c += 256) lmax = fmaxf(lmax, row[c]);
    red[tid] = lmax; __syncthreads();
    for (int s = 128; s > 0; s >>= 1) {
        if (tid < s) red[tid] = fmaxf(red[tid], red[tid + s]);
        __syncthreads();
    }
    float mx = red[0]; __syncthreads();

    float lsum = 0.0f;
    for (int c = tid; c < 1000; c += 256) lsum += __expf(row[c] - mx);
    red[tid] = lsum; __syncthreads();
    for (int s = 128; s > 0; s >>= 1) {
        if (tid < s) red[tid] += red[tid + s];
        __syncthreads();
    }
    float inv = 1.0f / red[0];

    for (int c = tid; c < 1024; c += 256) {
        float v = (c < 1000) ? __expf(row[c] - mx) * inv : 0.0f;
        __bf16 h = f2bf(v);
        dT_hi[(size_t)c * 128 + leaf] = h;
        dT_lo[(size_t)c * 128 + leaf] = f2bf(v - bf2f(h));
    }
}

// ============================================================
// Kernel C: GEMM1 (logits = xs @ W^T + b) -> sigmoid -> tree path -> pa hi/lo
// Block tile: 64(M) x 128(N=nodes). 8 waves: wave = 1 Mtile x 4 Ntiles.
// B tiles (128 nodes x 32 K, bf16 hi+lo) staged by TDM into double-buffered
// LDS; wave 0 drives the DMA, everyone syncs via tensorcnt + barrier.
// ============================================================
__global__ __launch_bounds__(256) void sdt_gemm1(const float* __restrict__ xs,
                                                 const __bf16* __restrict__ Whi,
                                                 const __bf16* __restrict__ Wlo,
                                                 const float* __restrict__ bias,
                                                 __bf16* __restrict__ pa_hi,
                                                 __bf16* __restrict__ pa_lo) {
    __shared__ float ps[64 * 129];              // [row][node], stride 129
#if HAVE_TDM
    // bbuf[buf][hi=0/lo=1][node*64B + k*2B]  (8 KB per quadrant)
    __shared__ __align__(128) char bbuf[2][2][128 * 64];
#endif
    const int tid  = threadIdx.x;
    const int w    = tid >> 5;                  // wave 0..7
    const int lane = tid & 31;
    const int l16  = lane & 15;
    const int h16  = lane >> 4;                 // 0: lanes 0-15, 1: lanes 16-31
    const int m0   = blockIdx.x * 64;
    const int wm   = (w >> 1) * 16;             // wave M offset (0/16/32/48)
    const int wn   = (w & 1) * 64;              // wave N offset (0/64)

    v8f acc[4];
#pragma unroll
    for (int i = 0; i < 4; ++i) acc[i] = zero8();

    const float* aptr = xs + (size_t)(m0 + wm + l16) * 2048;

#if HAVE_TDM
    const bool issuer = (w == 0);
    if (issuer) {
        tdm_load_2d_b16(Whi, lds_off_u32(&bbuf[0][0][0]), 32, 128, 2048, 128, 2048);
        tdm_load_2d_b16(Wlo, lds_off_u32(&bbuf[0][1][0]), 32, 128, 2048, 128, 2048);
        __builtin_amdgcn_s_wait_tensorcnt(0);
    }
    __syncthreads();
#endif

    for (int kb = 0; kb < 64; ++kb) {
        const int kbase = kb * 32;
#if HAVE_TDM
        const int cur = kb & 1;
        if (issuer && (kb + 1 < 64)) {
            tdm_load_2d_b16(Whi + (size_t)(kb + 1) * 32,
                            lds_off_u32(&bbuf[1 - cur][0][0]), 32, 128, 2048, 128, 2048);
            tdm_load_2d_b16(Wlo + (size_t)(kb + 1) * 32,
                            lds_off_u32(&bbuf[1 - cur][1][0]), 32, 128, 2048, 128, 2048);
        }
#endif
        if (kb < 63) __builtin_prefetch(aptr + kbase + 32, 0, 1);

        // ---- A fragment: row m0+wm+l16; elements 0..15 -> K=koff+e
        //      (koff = h16*16) ----
        BF16x16 ahi, alo;
        {
            const int koff = kbase + h16 * 16;
            f32x4 r0 = *(const f32x4*)(aptr + koff);
            f32x4 r1 = *(const f32x4*)(aptr + koff + 4);
            f32x4 r2 = *(const f32x4*)(aptr + koff + 8);
            f32x4 r3 = *(const f32x4*)(aptr + koff + 12);
            float xv[16];
#pragma unroll
            for (int i = 0; i < 4; ++i) {
                xv[i] = r0[i]; xv[4 + i] = r1[i];
                xv[8 + i] = r2[i]; xv[12 + i] = r3[i];
            }
#pragma unroll
            for (int e = 0; e < 16; ++e) {
                __bf16 h = f2bf(xv[e]);
                ahi.e[e] = h;
                alo.e[e] = f2bf(xv[e] - bf2f(h));
            }
        }

        // ---- B fragments: lane holds node n = wn+nt*16+l16,
        //      K = kbase + h16*16 + e (contiguous 16 bf16 = 32B) ----
        BF16x16 bhi[4], blo[4];
#pragma unroll
        for (int nt = 0; nt < 4; ++nt) {
            const int n = wn + nt * 16 + l16;
#if HAVE_TDM
            const char* ph = &bbuf[cur][0][n * 64 + h16 * 32];
            const char* pl = &bbuf[cur][1][n * 64 + h16 * 32];
            bhi[nt].h[0] = *(const bf16x8*)(ph);
            bhi[nt].h[1] = *(const bf16x8*)(ph + 16);
            blo[nt].h[0] = *(const bf16x8*)(pl);
            blo[nt].h[1] = *(const bf16x8*)(pl + 16);
#else
            const size_t off = (size_t)n * 2048 + kbase + h16 * 16;
            bhi[nt].h[0] = *(const bf16x8*)(Whi + off);
            bhi[nt].h[1] = *(const bf16x8*)(Whi + off + 8);
            blo[nt].h[0] = *(const bf16x8*)(Wlo + off);
            blo[nt].h[1] = *(const bf16x8*)(Wlo + off + 8);
#endif
        }

        // ---- 12 WMMAs: hi*hi + hi*lo + lo*hi ----
#pragma unroll
        for (int nt = 0; nt < 4; ++nt) {
            acc[nt] = wmma_bf16(ahi.v, bhi[nt].v, acc[nt]);
            acc[nt] = wmma_bf16(ahi.v, blo[nt].v, acc[nt]);
            acc[nt] = wmma_bf16(alo.v, bhi[nt].v, acc[nt]);
        }

#if HAVE_TDM
        if (issuer && (kb + 1 < 64)) __builtin_amdgcn_s_wait_tensorcnt(0);
        __syncthreads();
#endif
    }

    // ---- epilogue: bias + sigmoid -> LDS ----
#pragma unroll
    for (int nt = 0; nt < 4; ++nt) {
        const int c = wn + nt * 16 + l16;                 // node 0..127
        const float bb = (c < 127) ? bias[c] : 0.0f;
#pragma unroll
        for (int v = 0; v < 8; ++v) {
            const int r = wm + v + h16 * 8;               // row-in-tile
            float x = acc[nt][v] + bb;
            ps[r * 129 + c] = 1.0f / (1.0f + __expf(-x));
        }
    }
    __syncthreads();

    // ---- tree path products -> pa (bf16 hi/lo) ----
    for (int it = 0; it < 32; ++it) {
        int idx  = tid + it * 256;                        // 0..8191
        int row  = idx >> 7;                              // 0..63
        int leaf = idx & 127;
        float pa = 1.0f;
#pragma unroll
        for (int k = 0; k < 7; ++k) {
            int node = (1 << k) - 1 + (leaf >> (7 - k));
            int bit  = (leaf >> (6 - k)) & 1;
            float p  = ps[row * 129 + node];
            pa *= bit ? p : (1.0f - p);
        }
        __bf16 h = f2bf(pa);
        size_t o = (size_t)(m0 + row) * 128 + leaf;
        pa_hi[o] = h;
        pa_lo[o] = f2bf(pa - bf2f(h));
    }
}

// ============================================================
// Kernel D: out = pa @ distsT^T  (M=8192, K=128, N=1000)
// Grid: (128 M-blocks, 8 N-blocks). Same wave tiling as GEMM1.
// ============================================================
__global__ __launch_bounds__(256) void sdt_gemm2(const __bf16* __restrict__ pa_hi,
                                                 const __bf16* __restrict__ pa_lo,
                                                 const __bf16* __restrict__ dT_hi,
                                                 const __bf16* __restrict__ dT_lo,
                                                 float* __restrict__ out) {
    const int tid  = threadIdx.x;
    const int w    = tid >> 5;
    const int lane = tid & 31;
    const int l16  = lane & 15;
    const int h16  = lane >> 4;
    const int m0   = blockIdx.x * 64;
    const int n0   = blockIdx.y * 128;
    const int wm   = (w >> 1) * 16;
    const int wn   = (w & 1) * 64;

    v8f acc[4];
#pragma unroll
    for (int i = 0; i < 4; ++i) acc[i] = zero8();

    const __bf16* ah = pa_hi + (size_t)(m0 + wm + l16) * 128;
    const __bf16* al = pa_lo + (size_t)(m0 + wm + l16) * 128;

#pragma unroll
    for (int kb = 0; kb < 4; ++kb) {
        const int kbase = kb * 32;
        BF16x16 ahi, alo;
        ahi.h[0] = *(const bf16x8*)(ah + kbase + h16 * 16);
        ahi.h[1] = *(const bf16x8*)(ah + kbase + 8 + h16 * 16);
        alo.h[0] = *(const bf16x8*)(al + kbase + h16 * 16);
        alo.h[1] = *(const bf16x8*)(al + kbase + 8 + h16 * 16);

        BF16x16 bhi[4], blo[4];
#pragma unroll
        for (int nt = 0; nt < 4; ++nt) {
            const size_t off = (size_t)(n0 + wn + nt * 16 + l16) * 128 + kbase + h16 * 16;
            bhi[nt].h[0] = *(const bf16x8*)(dT_hi + off);
            bhi[nt].h[1] = *(const bf16x8*)(dT_hi + off + 8);
            blo[nt].h[0] = *(const bf16x8*)(dT_lo + off);
            blo[nt].h[1] = *(const bf16x8*)(dT_lo + off + 8);
        }
#pragma unroll
        for (int nt = 0; nt < 4; ++nt) {
            acc[nt] = wmma_bf16(ahi.v, bhi[nt].v, acc[nt]);
            acc[nt] = wmma_bf16(ahi.v, blo[nt].v, acc[nt]);
            acc[nt] = wmma_bf16(alo.v, bhi[nt].v, acc[nt]);
        }
    }

#pragma unroll
    for (int nt = 0; nt < 4; ++nt) {
        const int col = n0 + wn + nt * 16 + l16;
        if (col < 1000) {
#pragma unroll
            for (int v = 0; v < 8; ++v) {
                const int row = m0 + wm + v + h16 * 8;
                out[(size_t)row * 1000 + col] = acc[nt][v];
            }
        }
    }
}

// ============================================================
// Launcher
// inputs: xs(8192x2048 f32), W(127x2048 f32), b(127 f32), leaf_params(128x1000 f32)
// output: 8192x1000 f32
// ============================================================
extern "C" void kernel_launch(void* const* d_in, const int* in_sizes, int n_in,
                              void* d_out, int out_size, void* d_ws, size_t ws_size,
                              hipStream_t stream) {
    const float* xs = (const float*)d_in[0];
    const float* W  = (const float*)d_in[1];
    const float* b  = (const float*)d_in[2];
    const float* lp = (const float*)d_in[3];
    float* out = (float*)d_out;

    char* ws = (char*)d_ws;
    const size_t SZ_W  = (size_t)128 * 2048 * 2;   // 512 KB per array
    const size_t SZ_DT = (size_t)1024 * 128 * 2;   // 256 KB per array
    const size_t SZ_PA = (size_t)8192 * 128 * 2;   // 2 MB per array
    __bf16* Whi   = (__bf16*)(ws);
    __bf16* Wlo   = (__bf16*)(ws + SZ_W);
    __bf16* dT_hi = (__bf16*)(ws + 2 * SZ_W);
    __bf16* dT_lo = (__bf16*)(ws + 2 * SZ_W + SZ_DT);
    __bf16* pa_hi = (__bf16*)(ws + 2 * SZ_W + 2 * SZ_DT);
    __bf16* pa_lo = (__bf16*)(ws + 2 * SZ_W + 2 * SZ_DT + SZ_PA);

    sdt_wconv<<<(128 * 2048) / 256, 256, 0, stream>>>(W, Whi, Wlo);
    sdt_softmax_t<<<128, 256, 0, stream>>>(lp, dT_hi, dT_lo);
    sdt_gemm1<<<8192 / 64, 256, 0, stream>>>(xs, Whi, Wlo, b, pa_hi, pa_lo);
    sdt_gemm2<<<dim3(8192 / 64, 8), 256, 0, stream>>>(pa_hi, pa_lo, dT_hi, dT_lo, out);
}
